// CausalSelfAttention_11699490915056
// MI455X (gfx1250) — matrix operand order repaired and checked by reading, not compiled, not run on hardware
//
#include <hip/hip_runtime.h>
#include <hip/hip_bf16.h>

// ---------------------------------------------------------------------------
// Causal self-attention w/ ALiBi-like log1p decay bias for MI455X (gfx1250).
// B=4, T=2048, C=1024, H=16, D=64.  bf16 WMMA (v_wmma_f32_16x16x32_bf16),
// f32 accumulation, flash-attention (online softmax), 3-kernel pipeline.
// CDNA5 paths: v_wmma, global_load_async_to_lds_b128 (ASYNCcnt),
// global_prefetch for next K-slab.
// ---------------------------------------------------------------------------

#define B_   4
#define T_   2048
#define C_   1024
#define H_   16
#define D_   64
#define C3_  3072

typedef __attribute__((ext_vector_type(4)))  __bf16 v4bf;
typedef __attribute__((ext_vector_type(8)))  __bf16 v8bf;
typedef __attribute__((ext_vector_type(16))) __bf16 v16bf;
typedef __attribute__((ext_vector_type(4)))  float  v4f;
typedef __attribute__((ext_vector_type(8)))  float  v8f;

#define WMMA_BF16(a, b, c) \
  __builtin_amdgcn_wmma_f32_16x16x32_bf16(false, (a), false, (b), (short)0, (c), false, false)

static __device__ __forceinline__ v8f v8f_zero() {
  v8f z;
  for (int i = 0; i < 8; ++i) z[i] = 0.0f;
  return z;
}

static __device__ __forceinline__ v16bf cat8(v8bf lo, v8bf hi) {
  return __builtin_shufflevector(lo, hi, 0, 1, 2, 3, 4, 5, 6, 7,
                                         8, 9, 10, 11, 12, 13, 14, 15);
}

// Async global -> LDS copy of 16 bytes (GLOBAL_LOAD_ASYNC_TO_LDS_B128, GV
// mode: 64-bit vaddr, saddr=off). VDST VGPR carries the LDS byte address
// (low 32 bits of the generic shared pointer). Tracked on ASYNCcnt.
static __device__ __forceinline__ void async_copy_b128(void* lds_dst,
                                                       const void* gsrc) {
  unsigned l = (unsigned)(unsigned long long)lds_dst;
  asm volatile("global_load_async_to_lds_b128 %0, %1, off"
               :
               : "v"(l), "v"(gsrc)
               : "memory");
}

static __device__ __forceinline__ void wait_async0() {
  asm volatile("s_wait_asynccnt 0" ::: "memory");
}

// A fragment (16 x 32 bf16, M x K) from row-major LDS tile.
static __device__ __forceinline__ v16bf frag_a(const __bf16* tile, int row,
                                               int k0, int lane, int lda) {
  const __bf16* p = tile + row * lda + k0 + ((lane >> 4) << 3);
  v8bf lo = *(const v8bf*)p;
  v8bf hi = *(const v8bf*)(p + 16);
  return cat8(lo, hi);
}

// B fragment (32 x 16 bf16, K x N) from TRANSPOSED LDS tile Bt[n][k].
static __device__ __forceinline__ v16bf frag_b(const __bf16* tileT, int col,
                                               int k0, int lane, int ldb) {
  const __bf16* p = tileT + col * ldb + k0 + ((lane >> 4) << 3);
  v8bf lo = *(const v8bf*)p;
  v8bf hi = *(const v8bf*)(p + 16);
  return cat8(lo, hi);
}

// ---------------------------------------------------------------------------
// Kernel 1: qkv = x @ w_attn, scatter into q/k/v [B,H,T,D] bf16, q scaled 1/8.
// ---------------------------------------------------------------------------
__global__ __launch_bounds__(256)
void qkv_gemm_kernel(const float* __restrict__ x, const float* __restrict__ w,
                     __bf16* __restrict__ qb, __bf16* __restrict__ kb,
                     __bf16* __restrict__ vb) {
  const int LDA = 40;
  const int LDB = 40;
  __shared__ __bf16 As[128 * 40];   // A row-major [m][k]
  __shared__ __bf16 Bt[128 * 40];   // B transposed [n][k]

  const int tid = threadIdx.x;
  const int lane = tid & 31;
  const int wid = tid >> 5;
  const int wm = wid & 1;
  const int wn = wid >> 1;
  const int m0 = blockIdx.y * 128;
  const int n0 = blockIdx.x * 128;

  v8f acc[4][2];
  for (int mf = 0; mf < 4; ++mf)
    for (int nf = 0; nf < 2; ++nf) acc[mf][nf] = v8f_zero();

  for (int k0 = 0; k0 < C_; k0 += 32) {
    __syncthreads();
    // A tile: 128x32 f32 -> bf16
    for (int i = 0; i < 4; ++i) {
      int e = tid + i * 256;
      int r = e >> 3;
      int c4 = (e & 7) << 2;
      const float* src = x + (size_t)(m0 + r) * C_ + k0 + c4;
      v4f v = *(const v4f*)src;
      if (i == 0 && k0 + 32 < C_) __builtin_prefetch(src + 32, 0, 1);
      v4bf bv;
      for (int j = 0; j < 4; ++j) bv[j] = (__bf16)v[j];
      *(v4bf*)(As + r * LDA + c4) = bv;
    }
    // B tile: 32x128 f32 -> bf16, stored transposed [n][k]
    for (int i = 0; i < 4; ++i) {
      int e = tid + i * 256;
      int kr = e >> 5;
      int c4 = (e & 31) << 2;
      const float* src = w + (size_t)(k0 + kr) * C3_ + n0 + c4;
      v4f v = *(const v4f*)src;
      if (i == 0 && k0 + 32 < C_) __builtin_prefetch(src + (size_t)32 * C3_, 0, 1);
      for (int j = 0; j < 4; ++j) Bt[(c4 + j) * LDB + kr] = (__bf16)v[j];
    }
    __syncthreads();

    v16bf af[4], bfr[2];
    for (int mf = 0; mf < 4; ++mf)
      af[mf] = frag_a(As, wm * 64 + mf * 16 + (lane & 15), 0, lane, LDA);
    for (int nf = 0; nf < 2; ++nf)
      bfr[nf] = frag_b(Bt, wn * 32 + nf * 16 + (lane & 15), 0, lane, LDB);
    for (int mf = 0; mf < 4; ++mf)
      for (int nf = 0; nf < 2; ++nf)
        acc[mf][nf] = WMMA_BF16(af[mf], bfr[nf], acc[mf][nf]);
  }

  const int colc = lane & 15;
  const int rowb = (lane >> 4) << 3;
  for (int mf = 0; mf < 4; ++mf) {
    for (int nf = 0; nf < 2; ++nf) {
      int n = n0 + wn * 32 + nf * 16 + colc;
      int which = n >> 10;       // 0=q 1=k 2=v (uniform per frag)
      int hc = n & 1023;
      int h = hc >> 6;
      int d = hc & 63;
      __bf16* dst = (which == 0) ? qb : ((which == 1) ? kb : vb);
      float scl = (which == 0) ? 0.125f : 1.0f;
      for (int i = 0; i < 8; ++i) {
        int m = m0 + wm * 64 + mf * 16 + rowb + i;
        int b = m >> 11;
        int t = m & 2047;
        size_t idx = (((size_t)(b * H_ + h)) * T_ + t) * D_ + d;
        dst[idx] = (__bf16)(acc[mf][nf][i] * scl);
      }
    }
  }
}

// ---------------------------------------------------------------------------
// Kernel 2: flash attention per (b,h). Block = 128 threads (4 waves).
// Q/K tiles staged via GLOBAL_LOAD_ASYNC_TO_LDS_B128 (ASYNCcnt), V staged
// transposed through VGPRs (layout transform).
// ---------------------------------------------------------------------------
__global__ __launch_bounds__(128)
void flash_attn_kernel(const __bf16* __restrict__ qb,
                       const __bf16* __restrict__ kb,
                       const __bf16* __restrict__ vb,
                       const float* __restrict__ decay_raw,
                       __bf16* __restrict__ yb) {
  const int LDS = 72;
  __shared__ __bf16 Qs[64 * 72];  // [q][d]
  __shared__ __bf16 Ks[64 * 72];  // [key][d] (natural transposed-B layout)
  __shared__ __bf16 Vt[64 * 72];  // [d][key]
  __shared__ __bf16 Ps[64 * 72];  // [q][key] P staging (C/D -> A re-layout)

  const int tid = threadIdx.x;
  const int lane = tid & 31;
  const int wid = tid >> 5;
  const int colc = lane & 15;
  const int rowb = (lane >> 4) << 3;

  const int bh = blockIdx.y;
  const int b = bh >> 4;
  const int h = bh & 15;
  const int q0 = blockIdx.x * 64;

  const __bf16* qbh = qb + (size_t)bh * T_ * D_;
  const __bf16* kbh = kb + (size_t)bh * T_ * D_;
  const __bf16* vbh = vb + (size_t)bh * T_ * D_;

  const float decay = __logf(1.0f + __expf(decay_raw[h]));  // softplus

  // Q tile: async copy 64x64 bf16 straight into LDS
  for (int i = 0; i < 4; ++i) {
    int e = tid + i * 128;
    int r = e >> 3;
    int c8 = (e & 7) << 3;
    async_copy_b128(Qs + r * LDS + c8, qbh + (size_t)(q0 + r) * D_ + c8);
  }

  v8f o[4];
  for (int df = 0; df < 4; ++df) o[df] = v8f_zero();
  float mrun[8], lrun[8];
  for (int i = 0; i < 8; ++i) { mrun[i] = -1e30f; lrun[i] = 0.0f; }

  for (int j0 = 0; j0 <= q0 + 63; j0 += 64) {
    __syncthreads();
    // K tile row-major [key][d]: async copy
    for (int i = 0; i < 4; ++i) {
      int e = tid + i * 128;
      int r = e >> 3;
      int c8 = (e & 7) << 3;
      async_copy_b128(Ks + r * LDS + c8, kbh + (size_t)(j0 + r) * D_ + c8);
    }
    // V tile transposed [d][key]: through VGPRs
    for (int i = 0; i < 4; ++i) {
      int e = tid + i * 128;
      int r = e >> 3;
      int c8 = (e & 7) << 3;
      v8bf v = *(const v8bf*)(vbh + (size_t)(j0 + r) * D_ + c8);
      for (int j = 0; j < 8; ++j) Vt[(c8 + j) * LDS + r] = v[j];
    }
    wait_async0();
    __syncthreads();

    // S = Q @ K^T  (q pre-scaled by 1/sqrt(D) in kernel 1)
    v8f s[4];
    for (int nf = 0; nf < 4; ++nf) s[nf] = v8f_zero();
    for (int kk = 0; kk < D_; kk += 32) {
      v16bf aq = frag_a(Qs, wid * 16 + colc, kk, lane, LDS);
      for (int nf = 0; nf < 4; ++nf) {
        v16bf bk = frag_b(Ks, nf * 16 + colc, kk, lane, LDS);
        s[nf] = WMMA_BF16(aq, bk, s[nf]);
      }
    }

    // Causal mask + decay bias: -log1p(decay * log1p(dist))
    for (int nf = 0; nf < 4; ++nf) {
      int tk = j0 + nf * 16 + colc;
      for (int i = 0; i < 8; ++i) {
        int tq = q0 + wid * 16 + rowb + i;
        int dist = tq - tk;
        if (dist < 0) {
          s[nf][i] = -1e30f;
        } else {
          s[nf][i] -= __logf(1.0f + decay * __logf(1.0f + (float)dist));
        }
      }
    }

    // Online softmax update
    for (int i = 0; i < 8; ++i) {
      float rmax = fmaxf(fmaxf(s[0][i], s[1][i]), fmaxf(s[2][i], s[3][i]));
      rmax = fmaxf(rmax, __shfl_xor(rmax, 1));
      rmax = fmaxf(rmax, __shfl_xor(rmax, 2));
      rmax = fmaxf(rmax, __shfl_xor(rmax, 4));
      rmax = fmaxf(rmax, __shfl_xor(rmax, 8));
      float mnew = fmaxf(mrun[i], rmax);
      float scale = __expf(mrun[i] - mnew);
      float rsum = 0.0f;
      for (int nf = 0; nf < 4; ++nf) {
        float p = __expf(s[nf][i] - mnew);
        s[nf][i] = p;
        rsum += p;
      }
      rsum += __shfl_xor(rsum, 1);
      rsum += __shfl_xor(rsum, 2);
      rsum += __shfl_xor(rsum, 4);
      rsum += __shfl_xor(rsum, 8);
      lrun[i] = lrun[i] * scale + rsum;
      mrun[i] = mnew;
      for (int df = 0; df < 4; ++df) o[df][i] *= scale;
    }

    // Stage P (C/D layout) into LDS, re-read as A layout.
    for (int nf = 0; nf < 4; ++nf)
      for (int i = 0; i < 8; ++i)
        Ps[(wid * 16 + rowb + i) * LDS + nf * 16 + colc] = (__bf16)s[nf][i];
    __syncthreads();

    // O += P @ V
    const __bf16* Pw = Ps + wid * 16 * LDS;
    for (int kk = 0; kk < 64; kk += 32) {
      v16bf ap = frag_a(Pw, colc, kk, lane, LDS);
      for (int df = 0; df < 4; ++df) {
        v16bf bv = frag_b(Vt, df * 16 + colc, kk, lane, LDS);
        o[df] = WMMA_BF16(ap, bv, o[df]);
      }
    }
  }

  // Normalize and write y [B,T,C] bf16
  for (int i = 0; i < 8; ++i) {
    float inv = 1.0f / lrun[i];
    int tq = q0 + wid * 16 + rowb + i;
    for (int df = 0; df < 4; ++df) {
      int c = h * 64 + df * 16 + colc;
      yb[(size_t)(b * T_ + tq) * C_ + c] = (__bf16)(o[df][i] * inv);
    }
  }
}

// ---------------------------------------------------------------------------
// Kernel 3: out = y @ w_proj, f32 output. A tile staged async (bf16 direct).
// ---------------------------------------------------------------------------
__global__ __launch_bounds__(256)
void proj_gemm_kernel(const __bf16* __restrict__ y, const float* __restrict__ w,
                      float* __restrict__ out) {
  const int LDA = 40;
  const int LDB = 40;
  __shared__ __bf16 As[128 * 40];
  __shared__ __bf16 Bt[128 * 40];

  const int tid = threadIdx.x;
  const int lane = tid & 31;
  const int wid = tid >> 5;
  const int wm = wid & 1;
  const int wn = wid >> 1;
  const int m0 = blockIdx.y * 128;
  const int n0 = blockIdx.x * 128;

  v8f acc[4][2];
  for (int mf = 0; mf < 4; ++mf)
    for (int nf = 0; nf < 2; ++nf) acc[mf][nf] = v8f_zero();

  for (int k0 = 0; k0 < C_; k0 += 32) {
    __syncthreads();
    // A tile: 128x32 bf16, async copy (no conversion needed)
    for (int i = 0; i < 2; ++i) {
      int e = tid + i * 256;
      int r = e >> 2;
      int c8 = (e & 3) << 3;
      async_copy_b128(As + r * LDA + c8,
                      y + (size_t)(m0 + r) * C_ + k0 + c8);
    }
    // B tile: 32x128 f32 -> bf16, transposed
    for (int i = 0; i < 4; ++i) {
      int e = tid + i * 256;
      int kr = e >> 5;
      int c4 = (e & 31) << 2;
      const float* src = w + (size_t)(k0 + kr) * C_ + n0 + c4;
      v4f v = *(const v4f*)src;
      if (i == 0 && k0 + 32 < C_) __builtin_prefetch(src + (size_t)32 * C_, 0, 1);
      for (int j = 0; j < 4; ++j) Bt[(c4 + j) * LDB + kr] = (__bf16)v[j];
    }
    wait_async0();
    __syncthreads();

    v16bf af[4], bfr[2];
    for (int mf = 0; mf < 4; ++mf)
      af[mf] = frag_a(As, wm * 64 + mf * 16 + (lane & 15), 0, lane, LDA);
    for (int nf = 0; nf < 2; ++nf)
      bfr[nf] = frag_b(Bt, wn * 32 + nf * 16 + (lane & 15), 0, lane, LDB);
    for (int mf = 0; mf < 4; ++mf)
      for (int nf = 0; nf < 2; ++nf)
        acc[mf][nf] = WMMA_BF16(af[mf], bfr[nf], acc[mf][nf]);
  }

  const int colc = lane & 15;
  const int rowb = (lane >> 4) << 3;
  for (int mf = 0; mf < 4; ++mf) {
    for (int nf = 0; nf < 2; ++nf) {
      int n = n0 + wn * 32 + nf * 16 + colc;
      for (int i = 0; i < 8; ++i) {
        int m = m0 + wm * 64 + mf * 16 + rowb + i;
        out[(size_t)m * C_ + n] = acc[mf][nf][i];
      }
    }
  }
}

// ---------------------------------------------------------------------------
// Launch: ws layout = q(16MB) | k(16MB) | v(16MB) | y(16MB), all bf16.
// ---------------------------------------------------------------------------
extern "C" void kernel_launch(void* const* d_in, const int* in_sizes, int n_in,
                              void* d_out, int out_size, void* d_ws,
                              size_t ws_size, hipStream_t stream) {
  const float* x = (const float*)d_in[0];
  const float* w_attn = (const float*)d_in[1];
  const float* w_proj = (const float*)d_in[2];
  const float* decay_raw = (const float*)d_in[3];

  char* ws = (char*)d_ws;
  const size_t seg = (size_t)B_ * H_ * T_ * D_ * sizeof(__bf16);  // 16 MB
  __bf16* qb = (__bf16*)(ws);
  __bf16* kb = (__bf16*)(ws + seg);
  __bf16* vb = (__bf16*)(ws + 2 * seg);
  __bf16* yb = (__bf16*)(ws + 3 * seg);

  qkv_gemm_kernel<<<dim3(C3_ / 128, (B_ * T_) / 128), 256, 0, stream>>>(
      x, w_attn, qb, kb, vb);
  flash_attn_kernel<<<dim3(T_ / 64, B_ * H_), 128, 0, stream>>>(
      qb, kb, vb, decay_raw, yb);
  proj_gemm_kernel<<<dim3(C_ / 128, (B_ * T_) / 128), 256, 0, stream>>>(
      yb, w_proj, (float*)d_out);
}
